// MixtureOfAttention_39865886442148
// MI455X (gfx1250) — matrix operand order repaired and checked by reading, not compiled, not run on hardware
//
#include <hip/hip_runtime.h>

#define TOPK   2
#define HNUM   16
#define BATCH  8
#define NSEQ   2048
#define DHEAD  64
#define GDIM   128
#define BQ     64     // query rows per block (4 waves x 16)
#define KCHUNK 64     // keys per LDS-staged chunk
#define LSTR   72     // padded LDS stride (bf16 elems) against bank conflicts

typedef __attribute__((ext_vector_type(16))) __bf16 bf16x16;
typedef __attribute__((ext_vector_type(8)))  float  f32x8;
typedef __attribute__((ext_vector_type(4)))  float  f32x4;

union AFrag { unsigned short u[16]; bf16x16 v; };   // 256-bit WMMA A/B fragment
union Acc8  { float e[8];          f32x8  v; };     // 256-bit WMMA C/D accumulator

// round-to-nearest-even f32 -> bf16 (pure bit ops: deterministic codegen)
__device__ __forceinline__ unsigned short f2bf(float f) {
    unsigned u = __builtin_bit_cast(unsigned, f);
    u += 0x7fffu + ((u >> 16) & 1u);
    return (unsigned short)(u >> 16);
}

// ---------------------------------------------------------------------------
// Kernel 1: per-batch gating. colmean over N, 16 logits, softmax, top-2.
// ---------------------------------------------------------------------------
__global__ __launch_bounds__(256) void moa_gate_kernel(
    const float* __restrict__ gi, const float* __restrict__ W,
    const float* __restrict__ bg, float* __restrict__ gs, int* __restrict__ topidx)
{
    const int b = blockIdx.x, tid = threadIdx.x;
    __shared__ float red[256];
    __shared__ float colmean[GDIM];
    __shared__ float logits[HNUM];

    const int g = tid & (GDIM - 1), p = tid >> 7;
    const float* base = gi + (size_t)b * NSEQ * GDIM + g;
    float s = 0.0f;
    for (int n = p; n < NSEQ; n += 2) s += base[(size_t)n * GDIM];
    red[tid] = s;
    __syncthreads();
    if (p == 0) colmean[g] = (red[g] + red[g + GDIM]) * (1.0f / (float)NSEQ);
    __syncthreads();

    const int h = tid >> 4, part = tid & 15;
    float t = 0.0f;
    #pragma unroll
    for (int kk = 0; kk < 8; ++kk)
        t += colmean[part * 8 + kk] * W[h * GDIM + part * 8 + kk];
    red[tid] = t;
    __syncthreads();
    for (int off = 8; off > 0; off >>= 1) {
        if (part < off) red[tid] += red[tid + off];
        __syncthreads();
    }
    if (tid < HNUM) logits[tid] = red[tid * 16] + bg[tid];
    __syncthreads();

    if (tid == 0) {
        float mx = logits[0];
        #pragma unroll
        for (int i = 1; i < HNUM; ++i) mx = fmaxf(mx, logits[i]);
        float e[HNUM], den = 0.0f;
        #pragma unroll
        for (int i = 0; i < HNUM; ++i) { e[i] = __expf(logits[i] - mx); den += e[i]; }
        int i0 = 0;
        #pragma unroll
        for (int i = 1; i < HNUM; ++i) if (e[i] > e[i0]) i0 = i;
        int i1 = (i0 == 0) ? 1 : 0;
        #pragma unroll
        for (int i = 0; i < HNUM; ++i) if (i != i0 && e[i] > e[i1]) i1 = i;
        const float inv = 1.0f / den;
        #pragma unroll
        for (int i = 0; i < HNUM; ++i) gs[b * HNUM + i] = e[i] * inv;
        topidx[b * TOPK + 0] = i0;
        topidx[b * TOPK + 1] = i1;
    }
}

// ---------------------------------------------------------------------------
// Kernel 2: scalar load-balance loss -> d_out[B*N*128]
// ---------------------------------------------------------------------------
__global__ void moa_loss_kernel(const float* __restrict__ gs,
                                const int* __restrict__ topidx,
                                float* __restrict__ out_loss)
{
    if (threadIdx.x == 0 && blockIdx.x == 0) {
        float tokens[HNUM];
        #pragma unroll
        for (int i = 0; i < HNUM; ++i) tokens[i] = 0.0f;
        for (int b = 0; b < BATCH; ++b)
            for (int s = 0; s < TOPK; ++s) tokens[topidx[b * TOPK + s]] += 1.0f;
        float tot = 0.0f;
        #pragma unroll
        for (int i = 0; i < HNUM; ++i) tot += tokens[i];
        float lb = 0.0f;
        for (int hh = 0; hh < HNUM; ++hh) {
            float mp = 0.0f;
            for (int b = 0; b < BATCH; ++b) mp += gs[b * HNUM + hh];
            mp *= (1.0f / (float)BATCH);
            lb += mp * (tokens[hh] / (tot + 1e-6f));
        }
        out_loss[0] = (float)HNUM * lb;
    }
}

// ---------------------------------------------------------------------------
// Kernel 3: flash attention on the selected heads, bf16 WMMA 16x16x32.
// grid = (N/64, TOPK, B), block = 128 (4 waves; wave w owns q rows w*16..+15)
// 64-key chunks; softmax denominator via WMMA against a ones matrix.
// B-fragments for each contraction chunk are hoisted into distinct registers
// so ds_load_b128 latency overlaps WMMA issue (incremental dscnt waits).
// ---------------------------------------------------------------------------
__global__ __launch_bounds__(128) void moa_attn_kernel(
    const float* __restrict__ q, const float* __restrict__ k,
    const float* __restrict__ v, const float* __restrict__ mask,
    const int* __restrict__ topidx, float* __restrict__ out)
{
    __shared__ unsigned short Kb[KCHUNK * LSTR];   // [key][d]   bf16
    __shared__ unsigned short Vt[DHEAD * LSTR];    // [d][key]   bf16 transposed
    __shared__ unsigned short Ps[4][16 * LSTR];    // per-wave P tile [m][key]

    const int qt   = blockIdx.x;
    const int slot = blockIdx.y;
    const int b    = blockIdx.z;
    const int h    = topidx[b * TOPK + slot];
    const int q0   = qt * BQ;

    const int lane = threadIdx.x & 31;
    const int wave = threadIdx.x >> 5;
    const int half = lane >> 4;      // lane half selects M/K sub-blocks per ISA layout
    const int n16  = lane & 15;

    const size_t headoff = ((size_t)b * HNUM + h) * (size_t)NSEQ * DHEAD;
    const float* kbase = k + headoff;
    const float* vbase = v + headoff;

    // --- Q A-fragments (two K=32 chunks over D=64), 1/sqrt(D) folded in ---
    AFrag aq[2];
    {
        const float* qr = q + headoff + (size_t)(q0 + wave * 16 + n16) * DHEAD;
        #pragma unroll
        for (int c = 0; c < 2; ++c) {
            const int b1 = c * 32 + half * 16;
            const int b2 = c * 32 + half * 16 + 8;
            #pragma unroll
            for (int t = 0; t < 8; ++t) aq[c].u[t]     = f2bf(qr[b1 + t] * 0.125f);
            #pragma unroll
            for (int t = 0; t < 8; ++t) aq[c].u[8 + t] = f2bf(qr[b2 + t] * 0.125f);
        }
    }

    // B-fragment of all-ones (bf16 1.0) for the P row-sum WMMA
    AFrag onesb;
    #pragma unroll
    for (int t = 0; t < 16; ++t) onesb.u[t] = 0x3F80u;

    // online-softmax state: this lane-half holds rows m = half*8 + r
    float mrow[8], lrow[8];
    Acc8 accO[4];
    #pragma unroll
    for (int r = 0; r < 8; ++r) { mrow[r] = -1.0e30f; lrow[r] = 0.0f; }
    #pragma unroll
    for (int t = 0; t < 4; ++t)
        #pragma unroll
        for (int r = 0; r < 8; ++r) accO[t].e[r] = 0.0f;

    const float* mbase = mask + ((size_t)b * NSEQ + q0 + wave * 16 + half * 8) * NSEQ;

    const int ldrow = threadIdx.x >> 1;         // 0..63 key row
    const int lddst = (threadIdx.x & 1) * 32;   // d segment (32 floats = one 128B line)

    for (int kc = 0; kc < NSEQ; kc += KCHUNK) {
        __syncthreads();   // previous chunk fully consumed before overwrite
        // --- cooperative K/V chunk: global f32 -> bf16 LDS (V transposed) ---
        {
            const float* kr = kbase + (size_t)(kc + ldrow) * DHEAD + lddst;
            const float* vr = vbase + (size_t)(kc + ldrow) * DHEAD + lddst;
            if (kc + KCHUNK < NSEQ) {
                __builtin_prefetch(kr + KCHUNK * DHEAD, 0, 3);
                __builtin_prefetch(vr + KCHUNK * DHEAD, 0, 3);
            }
            float fk[32], fv[32];
            #pragma unroll
            for (int t = 0; t < 8; ++t) {
                *(f32x4*)&fk[t * 4] = *(const f32x4*)(kr + t * 4);
                *(f32x4*)&fv[t * 4] = *(const f32x4*)(vr + t * 4);
            }
            // K: pack bf16 pairs -> one b32 LDS store per pair
            #pragma unroll
            for (int t = 0; t < 16; ++t) {
                unsigned pk = (unsigned)f2bf(fk[2 * t]) |
                              ((unsigned)f2bf(fk[2 * t + 1]) << 16);
                *(unsigned*)&Kb[ldrow * LSTR + lddst + 2 * t] = pk;
            }
            // V: transposed scatter (b16 stores)
            #pragma unroll
            for (int t = 0; t < 32; ++t) Vt[(lddst + t) * LSTR + ldrow] = f2bf(fv[t]);
        }
        __syncthreads();

        // --- S[s] = (Q*scale) @ K^T for key sub-tiles s=0..3 ---
        // load all four B-fragments of a contraction chunk before the WMMAs
        Acc8 S[4];
        #pragma unroll
        for (int s = 0; s < 4; ++s)
            #pragma unroll
            for (int r = 0; r < 8; ++r) S[s].e[r] = 0.0f;
        #pragma unroll
        for (int c = 0; c < 2; ++c) {
            AFrag bk[4];
            #pragma unroll
            for (int s = 0; s < 4; ++s) {
                const int off = (s * 16 + n16) * LSTR + c * 32 + half * 16;
                #pragma unroll
                for (int t = 0; t < 16; ++t) bk[s].u[t] = Kb[off + t];
            }
            #pragma unroll
            for (int s = 0; s < 4; ++s)
                S[s].v = __builtin_amdgcn_wmma_f32_16x16x32_bf16(
                    false, aq[c].v, false, bk[s].v, (short)0, S[s].v, false, false);
        }

        // --- additive mask ---
        #pragma unroll
        for (int s = 0; s < 4; ++s)
            #pragma unroll
            for (int r = 0; r < 8; ++r)
                S[s].e[r] += mbase[(size_t)r * NSEQ + kc + s * 16 + n16];

        // --- online softmax; only the row-max needs a shuffle tree ---
        float cm[8];
        #pragma unroll
        for (int r = 0; r < 8; ++r)
            cm[r] = fmaxf(fmaxf(S[0].e[r], S[1].e[r]), fmaxf(S[2].e[r], S[3].e[r]));
        #pragma unroll
        for (int msk = 1; msk < 16; msk <<= 1)
            #pragma unroll
            for (int r = 0; r < 8; ++r) cm[r] = fmaxf(cm[r], __shfl_xor(cm[r], msk, 32));

        #pragma unroll
        for (int r = 0; r < 8; ++r) {
            const float mn = fmaxf(mrow[r], cm[r]);
            const float sc = __expf(mrow[r] - mn);
            mrow[r] = mn;
            #pragma unroll
            for (int s = 0; s < 4; ++s) S[s].e[r] = __expf(S[s].e[r] - mn);
            lrow[r] *= sc;
            #pragma unroll
            for (int t = 0; t < 4; ++t) accO[t].e[r] *= sc;
        }

        // --- P: C-layout -> A-layout via per-wave LDS (same-wave DS ordering) ---
        unsigned short* ps = &Ps[wave][0];
        #pragma unroll
        for (int s = 0; s < 4; ++s)
            #pragma unroll
            for (int r = 0; r < 8; ++r)
                ps[(half * 8 + r) * LSTR + s * 16 + n16] = f2bf(S[s].e[r]);

        AFrag ap[2];
        #pragma unroll
        for (int cc = 0; cc < 2; ++cc) {
            const int ro = n16 * LSTR + cc * 32;
            #pragma unroll
            for (int t = 0; t < 8; ++t) ap[cc].u[t]     = ps[ro + half * 16 + t];
            #pragma unroll
            for (int t = 0; t < 8; ++t) ap[cc].u[8 + t] = ps[ro + half * 16 + 8 + t];
        }

        // --- chunk row-sums via WMMA: rs = P @ ones (broadcast over lanes) ---
        Acc8 rs;
        #pragma unroll
        for (int r = 0; r < 8; ++r) rs.e[r] = 0.0f;
        #pragma unroll
        for (int cc = 0; cc < 2; ++cc)
            rs.v = __builtin_amdgcn_wmma_f32_16x16x32_bf16(
                false, ap[cc].v, false, onesb.v, (short)0, rs.v, false, false);
        #pragma unroll
        for (int r = 0; r < 8; ++r) lrow[r] += rs.e[r];

        // --- O += P @ V over four d column tiles ---
        #pragma unroll
        for (int cc = 0; cc < 2; ++cc) {
            AFrag bv[4];
            #pragma unroll
            for (int t = 0; t < 4; ++t) {
                const int off = (t * 16 + n16) * LSTR + cc * 32 + half * 16;
                #pragma unroll
                for (int u = 0; u < 16; ++u) bv[t].u[u] = Vt[off + u];
            }
            #pragma unroll
            for (int t = 0; t < 4; ++t)
                accO[t].v = __builtin_amdgcn_wmma_f32_16x16x32_bf16(
                    false, ap[cc].v, false, bv[t].v, (short)0, accO[t].v, false, false);
        }
    }

    // --- epilogue: out[b, n, slot*64 + d] = O / l ---
    float* ob = out + ((size_t)b * NSEQ + q0 + wave * 16 + half * 8) * (TOPK * DHEAD)
                    + slot * DHEAD + n16;
    #pragma unroll
    for (int t = 0; t < 4; ++t) {
        #pragma unroll
        for (int r = 0; r < 8; ++r)
            ob[(size_t)r * (TOPK * DHEAD) + t * 16] = accO[t].e[r] / lrow[r];
    }
}

// ---------------------------------------------------------------------------
extern "C" void kernel_launch(void* const* d_in, const int* in_sizes, int n_in,
                              void* d_out, int out_size, void* d_ws, size_t ws_size,
                              hipStream_t stream)
{
    (void)in_sizes; (void)n_in; (void)out_size; (void)ws_size;
    const float* q    = (const float*)d_in[0];
    const float* k    = (const float*)d_in[1];
    const float* v    = (const float*)d_in[2];
    const float* gi   = (const float*)d_in[3];
    const float* mask = (const float*)d_in[4];
    const float* W    = (const float*)d_in[5];
    const float* bg   = (const float*)d_in[6];
    float* out = (float*)d_out;

    float* gs     = (float*)d_ws;                                   // B*H floats
    int*   topidx = (int*)((char*)d_ws + BATCH * HNUM * sizeof(float));

    moa_gate_kernel<<<BATCH, 256, 0, stream>>>(gi, W, bg, gs, topidx);
    moa_loss_kernel<<<1, 32, 0, stream>>>(gs, topidx,
        out + (size_t)BATCH * NSEQ * TOPK * DHEAD);

    dim3 grid(NSEQ / BQ, TOPK, BATCH);   // 32 x 2 x 8 = 512 blocks
    moa_attn_kernel<<<grid, 128, 0, stream>>>(q, k, v, mask, topidx, out);
}
